// Arch17GraphEncoder_40922448396394
// MI455X (gfx1250) — compile-verified, matrix operand order checked by reading, not run on hardware
//
#include <hip/hip_runtime.h>
#include <hip/hip_bf16.h>
#include <math.h>
#include <stdint.h>

typedef __attribute__((ext_vector_type(16))) _Float16 v16h;
typedef __attribute__((ext_vector_type(8)))  _Float16 v8h;
typedef __attribute__((ext_vector_type(8)))  float    v8f;

#define H_DIM 256
#define NHEAD 4
#define L_GNN 4
#define T_LAYERS 4
#define KMAXC 10
#define MAXD 32
#define B_BATCH 64
#define M_SUBT 32
#define KNODE 32
#define S_SUB (B_BATCH*M_SUBT)      /* 2048 */
#define N_NODES (S_SUB*KNODE)       /* 65536 */
#define E_EDGES 131072
#define EDGES_PER_SUB (E_EDGES/S_SUB)  /* 64 */
#define GEMM_RT 32

static __device__ __forceinline__ v16h pack16(v8h a, v8h b){
  return __builtin_shufflevector(a, b, 0,1,2,3,4,5,6,7,8,9,10,11,12,13,14,15);
}
static __device__ __forceinline__ float finite_or_zero(float v){
  return (v - v == 0.0f) ? v : 0.0f;   // finite -> v ; inf/NaN -> 0
}
// CDNA5 async copy: global memory -> LDS, tracked by ASYNCcnt (no VGPR bounce)
static __device__ __forceinline__ void async_copy_b128(unsigned lds_addr, const void* gaddr){
  asm volatile("global_load_async_to_lds_b128 %0, %1, off"
               :: "v"(lds_addr), "v"((unsigned long long)(uintptr_t)gaddr)
               : "memory");
}

// ---------------------------------------------------------------------------
// Weight convert + transpose: W[K][N] f32 -> Wt[N][K] f16
__global__ void k_wconv(const float* __restrict__ W, _Float16* __restrict__ Wt,
                        int K, int N){
  int idx = blockIdx.x*256 + threadIdx.x;
  if (idx >= K*N) return;
  int k = idx / N, n = idx % N;
  Wt[(size_t)n*K + k] = (_Float16)W[idx];
}

// ---------------------------------------------------------------------------
// Initial node features: h = (atom + dist + relu(lp*W+b)) * valid
__global__ __launch_bounds__(256) void k_init(
    const int* __restrict__ xid, const int* __restrict__ nodes,
    const int* __restrict__ dist, const float* __restrict__ lp,
    const float* __restrict__ atom, const float* __restrict__ diste,
    const float* __restrict__ lw, const float* __restrict__ lb,
    _Float16* __restrict__ h, float* __restrict__ validf){
  int n = blockIdx.x;
  int c = threadIdx.x;
  float lpv = finite_or_zero(lp[n >> 5]);
  int d = dist[n]; d = d < 0 ? 0 : (d > MAXD ? MAXD : d);
  float pe = fmaxf(lpv*lw[c] + lb[c], 0.f);
  float vf = (nodes[n] >= 0) ? 1.f : 0.f;
  if (c == 0) validf[n] = vf;
  h[(size_t)n*256 + c] =
      (_Float16)((atom[(size_t)xid[n]*256 + c] + diste[(size_t)d*256 + c] + pe) * vf);
}

// ---------------------------------------------------------------------------
// GIN aggregate per 32-node subgraph: z = (1+eps)*h + sum_dst relu(h[src]+e)
__global__ __launch_bounds__(256) void k_aggregate(
    const _Float16* __restrict__ h, const int* __restrict__ src,
    const int* __restrict__ dst, const int* __restrict__ eid,
    const float* __restrict__ bond, const float* __restrict__ epsArr, int layer,
    _Float16* __restrict__ z){
  __shared__ float hs[32][256];
  __shared__ float agg[32][256];
  int s = blockIdx.x, base = s*KNODE, eb = s*EDGES_PER_SUB, tid = threadIdx.x;
  for (int j = tid; j < 32*256; j += 256){
    int r = j >> 8, c = j & 255;
    hs[r][c] = (float)h[(size_t)(base + r)*256 + c];
    agg[r][c] = 0.f;
  }
  __syncthreads();
  int c = tid;  // each thread owns one feature column -> race-free
  for (int e = 0; e < EDGES_PER_SUB; ++e){
    int ei = eb + e;
    int sl = src[ei] - base, dl = dst[ei] - base, bi = eid[ei];
    float m = hs[sl][c] + bond[(size_t)bi*256 + c];
    agg[dl][c] += fmaxf(m, 0.f);
  }
  __syncthreads();
  float ep = 1.0f + epsArr[layer];
  for (int j = tid; j < 32*256; j += 256){
    int r = j >> 8, cc = j & 255;
    z[(size_t)(base + r)*256 + cc] = (_Float16)(ep*hs[r][cc] + agg[r][cc]);
  }
}

// ---------------------------------------------------------------------------
// Big GEMM, K fixed = 256. A[N][256] f16, Wt[256][256] f16 (pre-transposed).
// 16 waves; each wave owns 16 output cols and caches the full 256-deep B
// column block in registers. A tiles are double-buffered in LDS and staged
// with CDNA5 async global->LDS copies (ASYNCcnt).
// MODE 0: out = relu(acc+bias) f16 ; MODE 1: out = (acc+bias)*validf f16
template<int MODE>
__global__ __launch_bounds__(512) void k_gemm256(
    const _Float16* __restrict__ A, const _Float16* __restrict__ Wt,
    const float* __restrict__ bias, const float* __restrict__ validf,
    _Float16* __restrict__ out){
  __shared__ _Float16 sA[2][16*264];
  const int lane = threadIdx.x & 31;
  const int wave = threadIdx.x >> 5;
  const int n0   = wave*16;
  const int ncol = n0 + (lane & 15);
  const int hi   = lane >> 4;
  const int koff = hi*8;
  const int arow = lane & 15;

  // cache the whole 256-deep B column block (8 fragments, 64 VGPRs)
  v16h Bf[8];
  const _Float16* wrow = Wt + (size_t)ncol*256;
#pragma unroll
  for (int kk = 0; kk < 8; ++kk){
    v8h b0 = *(const v8h*)(wrow + kk*32 + koff);
    v8h b1 = *(const v8h*)(wrow + kk*32 + 16 + koff);
    Bf[kk] = pack16(b0, b1);
  }
  float bv = bias[ncol];

  // per-thread async-copy slot: one 16B chunk of the 16x256 A tile
  const unsigned ldsDst0 = (unsigned)(uintptr_t)&sA[0][wave*264 + lane*8];
  const unsigned ldsDst1 = (unsigned)(uintptr_t)&sA[1][wave*264 + lane*8];
  const _Float16* aBase = A + (size_t)blockIdx.x*GEMM_RT*16*256
                            + (size_t)wave*256 + lane*8;

  async_copy_b128(ldsDst0, aBase);                       // tile 0 in flight
  for (int rt = 0; rt < GEMM_RT; ++rt){
    const int cur = rt & 1;
    if (rt + 1 < GEMM_RT){
      async_copy_b128(cur ? ldsDst0 : ldsDst1, aBase + (size_t)(rt + 1)*16*256);
      asm volatile("s_wait_asynccnt 0x1" ::: "memory");  // tile rt landed
    } else {
      asm volatile("s_wait_asynccnt 0x0" ::: "memory");
    }
    __syncthreads();

    // batch all A-fragment LDS loads, then run the WMMA chain
    v16h Af[8];
    const _Float16* sa = &sA[cur][arow*264];
#pragma unroll
    for (int kk = 0; kk < 8; ++kk){
      v8h a0 = *(const v8h*)(sa + kk*32 + koff);
      v8h a1 = *(const v8h*)(sa + kk*32 + 16 + koff);
      Af[kk] = pack16(a0, a1);
    }
    v8f acc = {};
#pragma unroll
    for (int kk = 0; kk < 8; ++kk)
      acc = __builtin_amdgcn_wmma_f32_16x16x32_f16(
          false, Af[kk], false, Bf[kk], (short)0, acc, false, false);

    const int r0 = (blockIdx.x*GEMM_RT + rt)*16;
    if (MODE == 0){
#pragma unroll
      for (int i = 0; i < 8; ++i)
        out[(size_t)(r0 + 8*hi + i)*256 + ncol] = (_Float16)fmaxf(acc[i] + bv, 0.f);
    } else {
      float4 v0 = *(const float4*)(validf + r0 + 8*hi);
      float4 v1 = *(const float4*)(validf + r0 + 8*hi + 4);
      float vf[8] = {v0.x, v0.y, v0.z, v0.w, v1.x, v1.y, v1.z, v1.w};
#pragma unroll
      for (int i = 0; i < 8; ++i)
        out[(size_t)(r0 + 8*hi + i)*256 + ncol] = (_Float16)((acc[i] + bv)*vf[i]);
    }
    __syncthreads();   // protect sA[cur] before next iteration's async issue
  }
}

// ---------------------------------------------------------------------------
// Generic WMMA GEMM for transformer. A[M][K] f16, Wt[Nout][K] f16.
// 8 waves cover 128 output cols per block; grid = (M/16, Nout/128).
// MODE 0: f16 = acc+bias ; MODE 1: f16 = gelu(acc+bias) ; MODE 2: f32 += acc+bias
template<int MODE>
__global__ __launch_bounds__(256) void k_gemmgen(
    const _Float16* __restrict__ A, int K, const _Float16* __restrict__ Wt,
    int Nout, const float* __restrict__ bias, void* __restrict__ outp){
  __shared__ _Float16 sA[16*1032];
  const int lane = threadIdx.x & 31;
  const int wave = threadIdx.x >> 5;
  const int strideA = K + 8;
  const int r0 = blockIdx.x*16;
  const int cpr = K >> 3;  // 16B chunks per row
  for (int c = threadIdx.x; c < 16*cpr; c += 256){
    int row = c / cpr, seg = c % cpr;
    *(v8h*)(&sA[row*strideA + seg*8]) =
        *(const v8h*)(A + (size_t)(r0 + row)*K + seg*8);
  }
  __syncthreads();

  const int n0   = blockIdx.y*128 + wave*16;
  const int ncol = n0 + (lane & 15);
  const int hi   = lane >> 4;
  const int koff = hi*8;
  const int arow = lane & 15;
  const _Float16* wrow = Wt + (size_t)ncol*K;

  v8f acc = {};
  for (int kb = 0; kb < K; kb += 256){
    v16h Af[8], Bf[8];
#pragma unroll
    for (int kk = 0; kk < 8; ++kk){
      int k0 = kb + kk*32;
      v8h a0 = *(const v8h*)(&sA[arow*strideA + k0 + koff]);
      v8h a1 = *(const v8h*)(&sA[arow*strideA + k0 + 16 + koff]);
      Af[kk] = pack16(a0, a1);
      v8h b0 = *(const v8h*)(wrow + k0 + koff);
      v8h b1 = *(const v8h*)(wrow + k0 + 16 + koff);
      Bf[kk] = pack16(b0, b1);
    }
#pragma unroll
    for (int kk = 0; kk < 8; ++kk)
      acc = __builtin_amdgcn_wmma_f32_16x16x32_f16(
          false, Af[kk], false, Bf[kk], (short)0, acc, false, false);
  }
  float bv = bias ? bias[ncol] : 0.f;
#pragma unroll
  for (int i = 0; i < 8; ++i){
    size_t r = (size_t)r0 + 8*hi + i;
    float v = acc[i] + bv;
    if (MODE == 0){
      ((_Float16*)outp)[r*Nout + ncol] = (_Float16)v;
    } else if (MODE == 1){
      float g = 0.5f*v*(1.0f + erff(v*0.70710678118654752f));
      ((_Float16*)outp)[r*Nout + ncol] = (_Float16)g;
    } else {
      ((float*)outp)[r*Nout + ncol] += v;
    }
  }
}

// ---------------------------------------------------------------------------
// Gather root nodes: x[s][c] = h[s*KNODE][c]
__global__ __launch_bounds__(256) void k_gather(const _Float16* __restrict__ h,
                                                float* __restrict__ x){
  int r = blockIdx.x, c = threadIdx.x;
  x[(size_t)r*256 + c] = (float)h[(size_t)r*KNODE*256 + c];
}

// ---------------------------------------------------------------------------
// Overlap + attention bias per batch b: bias[b][i][j]
__global__ __launch_bounds__(1024) void k_bias(
    const int* __restrict__ nodes, const float* __restrict__ lp,
    const float* __restrict__ oemb, const float* __restrict__ alpha,
    float* __restrict__ bias){
  __shared__ int nd[32][32];
  __shared__ float lpc[32];
  int b = blockIdx.x, tid = threadIdx.x;
  nd[tid >> 5][tid & 31] = nodes[(size_t)b*1024 + tid];
  if (tid < 32){
    float v = finite_or_zero(lp[b*32 + tid]);
    lpc[tid] = fminf(fmaxf(v, -30.f), 0.f);
  }
  __syncthreads();
  int i = tid >> 5, j = tid & 31;
  int cnt = 0;
  for (int kk = 0; kk < 32; ++kk){
    int a = nd[i][kk];
    if (a >= 0){
      bool m = false;
      for (int t = 0; t < 32; ++t) m = m || (a == nd[j][t]);
      cnt += m ? 1 : 0;
    }
  }
  int o = cnt > KMAXC ? KMAXC : cnt;
  bias[(size_t)b*1024 + tid] = oemb[o] - alpha[0]*lpc[j];
}

// ---------------------------------------------------------------------------
// LayerNorm over H=256, f32 in -> f16 out
__global__ __launch_bounds__(256) void k_ln(
    const float* __restrict__ x, const float* __restrict__ g,
    const float* __restrict__ bb, _Float16* __restrict__ o16){
  __shared__ float red[256];
  int r = blockIdx.x, c = threadIdx.x;
  float v = x[(size_t)r*256 + c];
  red[c] = v; __syncthreads();
  for (int s2 = 128; s2 > 0; s2 >>= 1){ if (c < s2) red[c] += red[c + s2]; __syncthreads(); }
  float mu = red[0]*(1.f/256.f);
  __syncthreads();
  float dv = v - mu;
  red[c] = dv*dv; __syncthreads();
  for (int s2 = 128; s2 > 0; s2 >>= 1){ if (c < s2) red[c] += red[c + s2]; __syncthreads(); }
  float var = red[0]*(1.f/256.f);
  float y = dv*rsqrtf(var + 1e-5f)*g[c] + bb[c];
  o16[(size_t)r*256 + c] = (_Float16)y;
}

// ---------------------------------------------------------------------------
// Attention per (batch, head): 32x32 scores with additive bias, softmax, AV
__global__ __launch_bounds__(32) void k_attn(
    const _Float16* __restrict__ qkv, const float* __restrict__ bias,
    _Float16* __restrict__ o){
  __shared__ float ks[32][64];
  __shared__ float vs[32][64];
  int b = blockIdx.x >> 2, hh = blockIdx.x & 3;
  int tid = threadIdx.x;
  for (int j = tid; j < 2048; j += 32){
    int row = j >> 6, d = j & 63;
    size_t p = (size_t)(b*32 + row)*768 + hh*64 + d;
    ks[row][d] = (float)qkv[p + 256];
    vs[row][d] = (float)qkv[p + 512];
  }
  __syncthreads();
  int i = tid;
  float q[64];
  size_t qb = (size_t)(b*32 + i)*768 + hh*64;
#pragma unroll
  for (int d = 0; d < 64; ++d) q[d] = (float)qkv[qb + d];
  float sc[32]; float mx = -1e30f;
#pragma unroll
  for (int j = 0; j < 32; ++j){
    float s = 0.f;
#pragma unroll
    for (int d = 0; d < 64; ++d) s += q[d]*ks[j][d];
    s = s*0.125f + bias[(size_t)b*1024 + i*32 + j];
    sc[j] = s; mx = fmaxf(mx, s);
  }
  float sum = 0.f;
#pragma unroll
  for (int j = 0; j < 32; ++j){ float e = expf(sc[j] - mx); sc[j] = e; sum += e; }
  float inv = 1.f/sum;
  float accv[64];
#pragma unroll
  for (int d = 0; d < 64; ++d) accv[d] = 0.f;
#pragma unroll
  for (int j = 0; j < 32; ++j){
    float w = sc[j]*inv;
#pragma unroll
    for (int d = 0; d < 64; ++d) accv[d] += w*vs[j][d];
  }
  size_t ob = (size_t)(b*32 + i)*256 + hh*64;
#pragma unroll
  for (int d = 0; d < 64; ++d) o[ob + d] = (_Float16)accv[d];
}

// ---------------------------------------------------------------------------
// Weighted pooling: w = softmax(-clip(lp,-30,0)); out[b] = sum_i w_i * x_i
__global__ __launch_bounds__(256) void k_pool(
    const _Float16* __restrict__ xln, const float* __restrict__ lp,
    float* __restrict__ out){
  __shared__ float w[32];
  int b = blockIdx.x, tid = threadIdx.x;
  if (tid < 32){
    float v = finite_or_zero(lp[b*32 + tid]);
    w[tid] = -fminf(fmaxf(v, -30.f), 0.f);
  }
  __syncthreads();
  if (tid == 0){
    float mx = w[0];
    for (int i = 1; i < 32; ++i) mx = fmaxf(mx, w[i]);
    float s = 0.f;
    for (int i = 0; i < 32; ++i){ w[i] = expf(w[i] - mx); s += w[i]; }
    float inv = 1.f/s;
    for (int i = 0; i < 32; ++i) w[i] *= inv;
  }
  __syncthreads();
  int c = tid;
  float a = 0.f;
  for (int i = 0; i < 32; ++i) a += w[i]*(float)xln[(size_t)(b*32 + i)*256 + c];
  out[(size_t)b*256 + c] = a;
}

// ---------------------------------------------------------------------------
extern "C" void kernel_launch(void* const* d_in, const int* in_sizes, int n_in,
                              void* d_out, int out_size, void* d_ws, size_t ws_size,
                              hipStream_t stream){
  const int*   x_ids    = (const int*)d_in[0];
  const int*   edge_ids = (const int*)d_in[1];
  const int*   srcI     = (const int*)d_in[2];
  const int*   dstI     = (const int*)d_in[3];
  const int*   nodes    = (const int*)d_in[4];
  const int*   dist     = (const int*)d_in[5];
  const float* lp       = (const float*)d_in[6];
  const float* atom_emb = (const float*)d_in[7];
  const float* bond_emb = (const float*)d_in[8];
  const float* dist_emb = (const float*)d_in[9];
  const float* logp_W   = (const float*)d_in[10];
  const float* logp_b   = (const float*)d_in[11];
  const float* gnn_eps  = (const float*)d_in[12];
  const float* gnn_W1   = (const float*)d_in[13];
  const float* gnn_b1   = (const float*)d_in[14];
  const float* gnn_W2   = (const float*)d_in[15];
  const float* gnn_b2   = (const float*)d_in[16];
  const float* ln1_g    = (const float*)d_in[17];
  const float* ln1_b    = (const float*)d_in[18];
  const float* qkv_W    = (const float*)d_in[19];
  const float* out_W    = (const float*)d_in[20];
  const float* out_b    = (const float*)d_in[21];
  const float* ln2_g    = (const float*)d_in[22];
  const float* ln2_b    = (const float*)d_in[23];
  const float* f1_W     = (const float*)d_in[24];
  const float* f1_b     = (const float*)d_in[25];
  const float* f2_W     = (const float*)d_in[26];
  const float* f2_b     = (const float*)d_in[27];
  const float* lnout_g  = (const float*)d_in[28];
  const float* lnout_b  = (const float*)d_in[29];
  const float* overlapE = (const float*)d_in[30];
  const float* ht_alpha = (const float*)d_in[31];
  (void)in_sizes; (void)n_in; (void)out_size; (void)ws_size;

  char* ws = (char*)d_ws;
  size_t off = 0;
  auto alloc = [&](size_t bytes)->char*{
    char* p = ws + off;
    off += (bytes + 255) & ~(size_t)255;
    return p;
  };
  _Float16* hbuf  = (_Float16*)alloc((size_t)N_NODES*256*2);
  _Float16* zbuf  = (_Float16*)alloc((size_t)N_NODES*256*2);
  _Float16* ybuf  = (_Float16*)alloc((size_t)N_NODES*256*2);
  float*    validf= (float*)   alloc((size_t)N_NODES*4);
  _Float16* W1t   = (_Float16*)alloc((size_t)L_GNN*256*256*2);
  _Float16* W2t   = (_Float16*)alloc((size_t)L_GNN*256*256*2);
  _Float16* qkvWt = (_Float16*)alloc((size_t)T_LAYERS*768*256*2);
  _Float16* outWt = (_Float16*)alloc((size_t)T_LAYERS*256*256*2);
  _Float16* f1Wt  = (_Float16*)alloc((size_t)T_LAYERS*1024*256*2);
  _Float16* f2Wt  = (_Float16*)alloc((size_t)T_LAYERS*256*1024*2);
  float*    xbuf  = (float*)   alloc((size_t)S_SUB*256*4);
  _Float16* rbuf  = (_Float16*)alloc((size_t)S_SUB*256*2);
  _Float16* qkvb  = (_Float16*)alloc((size_t)S_SUB*768*2);
  _Float16* obuf  = (_Float16*)alloc((size_t)S_SUB*256*2);
  _Float16* ffh   = (_Float16*)alloc((size_t)S_SUB*1024*2);
  float*    biasb = (float*)   alloc((size_t)B_BATCH*1024*4);
  _Float16* xln   = (_Float16*)alloc((size_t)S_SUB*256*2);

  // --- weight conversion / transpose to f16 [Nout][K] ---
  for (int l = 0; l < L_GNN; ++l){
    k_wconv<<<(256*256 + 255)/256, 256, 0, stream>>>(gnn_W1 + (size_t)l*65536, W1t + (size_t)l*65536, 256, 256);
    k_wconv<<<(256*256 + 255)/256, 256, 0, stream>>>(gnn_W2 + (size_t)l*65536, W2t + (size_t)l*65536, 256, 256);
  }
  for (int t = 0; t < T_LAYERS; ++t){
    k_wconv<<<(256*768 + 255)/256, 256, 0, stream>>>(qkv_W + (size_t)t*256*768, qkvWt + (size_t)t*768*256, 256, 768);
    k_wconv<<<(256*256 + 255)/256, 256, 0, stream>>>(out_W + (size_t)t*65536,   outWt + (size_t)t*65536,   256, 256);
    k_wconv<<<(256*1024 + 255)/256, 256, 0, stream>>>(f1_W + (size_t)t*256*1024, f1Wt + (size_t)t*1024*256, 256, 1024);
    k_wconv<<<(1024*256 + 255)/256, 256, 0, stream>>>(f2_W + (size_t)t*1024*256, f2Wt + (size_t)t*256*1024, 1024, 256);
  }

  // --- initial features ---
  k_init<<<N_NODES, 256, 0, stream>>>(x_ids, nodes, dist, lp, atom_emb, dist_emb,
                                      logp_W, logp_b, hbuf, validf);

  // --- GNN layers (aggregate -> WMMA GEMM1 relu -> WMMA GEMM2 * valid) ---
  for (int l = 0; l < L_GNN; ++l){
    k_aggregate<<<S_SUB, 256, 0, stream>>>(hbuf, srcI, dstI, edge_ids, bond_emb,
                                           gnn_eps, l, zbuf);
    k_gemm256<0><<<N_NODES/(16*GEMM_RT), 512, 0, stream>>>(
        zbuf, W1t + (size_t)l*65536, gnn_b1 + l*256, nullptr, ybuf);
    k_gemm256<1><<<N_NODES/(16*GEMM_RT), 512, 0, stream>>>(
        ybuf, W2t + (size_t)l*65536, gnn_b2 + l*256, validf, hbuf);
  }

  // --- pool roots + attention bias ---
  k_gather<<<S_SUB, 256, 0, stream>>>(hbuf, xbuf);
  k_bias<<<B_BATCH, 1024, 0, stream>>>(nodes, lp, overlapE, ht_alpha, biasb);

  // --- transformer layers ---
  for (int t = 0; t < T_LAYERS; ++t){
    k_ln<<<S_SUB, 256, 0, stream>>>(xbuf, ln1_g + t*256, ln1_b + t*256, rbuf);
    k_gemmgen<0><<<dim3(S_SUB/16, 6), 256, 0, stream>>>(rbuf, 256, qkvWt + (size_t)t*768*256, 768, nullptr, qkvb);
    k_attn<<<B_BATCH*NHEAD, 32, 0, stream>>>(qkvb, biasb, obuf);
    k_gemmgen<2><<<dim3(S_SUB/16, 2), 256, 0, stream>>>(obuf, 256, outWt + (size_t)t*65536, 256, out_b + t*256, xbuf);
    k_ln<<<S_SUB, 256, 0, stream>>>(xbuf, ln2_g + t*256, ln2_b + t*256, rbuf);
    k_gemmgen<1><<<dim3(S_SUB/16, 8), 256, 0, stream>>>(rbuf, 256, f1Wt + (size_t)t*1024*256, 1024, f1_b + t*1024, ffh);
    k_gemmgen<2><<<dim3(S_SUB/16, 2), 256, 0, stream>>>(ffh, 1024, f2Wt + (size_t)t*256*1024, 256, f2_b + t*256, xbuf);
  }

  // --- final LN + lp-softmax pooling ---
  k_ln<<<S_SUB, 256, 0, stream>>>(xbuf, lnout_g, lnout_b, xln);
  k_pool<<<B_BATCH, 256, 0, stream>>>(xln, lp, (float*)d_out);
}
